// SelfAttention_5763846111327
// MI455X (gfx1250) — compile-verified, matrix operand order checked by reading, not run on hardware
//
#include <hip/hip_runtime.h>
#include <cstddef>
#include <cstdint>

typedef unsigned short u16;
typedef __attribute__((ext_vector_type(4)))  unsigned int uv4;
typedef __attribute__((ext_vector_type(4)))  float        fv4;
typedef __attribute__((ext_vector_type(16))) __bf16       v16bf;
typedef __attribute__((ext_vector_type(8)))  float        v8f;
typedef __attribute__((ext_vector_type(4)))  unsigned int u32x4;
typedef __attribute__((ext_vector_type(8)))  int          i32x8;
typedef __attribute__((ext_vector_type(4)))  int          i32x4;

#define S_LEN 1024
#define D_DIM 4096
#define H_NUM 64
#define HD    64
#define B_NUM 2

// float -> bf16 round-to-nearest-even
__device__ __forceinline__ u16 f2bf(float f) {
  unsigned int u = __float_as_uint(f);
  u += 0x7FFFu + ((u >> 16) & 1u);
  return (u16)(u >> 16);
}

// Load a 16x32 bf16 WMMA A-fragment (or 32x16 B-fragment, same lane pattern:
// lane&15 selects row/col, lane>>4 selects the K-half) from a row-major
// bf16 buffer with `stride` elements per row. Two 16-byte loads per lane.
__device__ __forceinline__ v16bf load_frag(const u16* base, int stride, int lane) {
  const int r  = lane & 15;
  const int hf = (lane >> 4) & 1;
  const u16* p = base + r * stride + hf * 8;
  struct U { uv4 a, b; } u;
  u.a = *(const uv4*)p;         // K = 8*hf .. 8*hf+7
  u.b = *(const uv4*)(p + 16);  // K = 16+8*hf .. 16+8*hf+7
  return __builtin_bit_cast(v16bf, u);
}

__device__ __forceinline__ v8f wmma_bf16(v16bf a, v16bf b, v8f c) {
  return __builtin_amdgcn_wmma_f32_16x16x32_bf16(false, a, false, b, (short)0, c, false, false);
}

// ---------------------------------------------------------------- TDM tile DMA
// Issue a TENSOR_LOAD_TO_LDS of a 2-D bf16 tile (tileCols x tileRows, row
// stride rowStride elements) from global `gtile` into LDS at byte offset
// `lds_off`, packed densely row-major in LDS. D# layout per CDNA5 ISA ch.8.
// Whole-wave instruction (EXEC ignored); tracked on TENSORcnt.
__device__ __forceinline__ void tdm_load_tile_2d(const u16* gtile, unsigned lds_off,
                                                 unsigned tileCols, unsigned tileRows,
                                                 unsigned rowStride) {
  const unsigned long long ga = (unsigned long long)(uintptr_t)gtile;
  u32x4 g0;
  g0[0] = 1u;                                                 // count=1, user mode
  g0[1] = lds_off;                                            // lds_addr (bytes)
  g0[2] = (unsigned)ga;                                       // global_addr[31:0]
  g0[3] = (unsigned)((ga >> 32) & 0x01FFFFFFu) | (2u << 30);  // addr[56:32] | type=2
  i32x8 g1;
  g1[0] = 1 << 16;                                            // data_size=1 -> 2B elems
  g1[1] = (int)((rowStride & 0xFFFFu) << 16);                 // tensor_dim0[15:0]
  g1[2] = (int)(((rowStride >> 16) & 0xFFFFu) | (tileRows << 16)); // dim0 hi | dim1 lo
  g1[3] = (int)(tileCols << 16);                              // dim1 hi(0) | tile_dim0
  g1[4] = (int)tileRows;                                      // tile_dim1 | tile_dim2=0
  g1[5] = (int)rowStride;                                     // tensor_dim0_stride lo
  g1[6] = 0;                                                  // stride hi | dim1_stride lo
  g1[7] = 0;
  const i32x4 z4 = (i32x4){0, 0, 0, 0};
#if defined(__clang_major__) && (__clang_major__ >= 23)
  const i32x8 z8 = (i32x8){0, 0, 0, 0, 0, 0, 0, 0};
  __builtin_amdgcn_tensor_load_to_lds(g0, g1, z4, z4, z8, 0);
#else
  __builtin_amdgcn_tensor_load_to_lds(g0, g1, z4, z4, 0);
#endif
}

// ---------------------------------------------------------------- weights cast
__global__ __launch_bounds__(256)
void cast_weights_kernel(const float* w0, const float* w1, const float* w2, const float* w3,
                         u16* o0, u16* o1, u16* o2, u16* o3) {
  const int z = blockIdx.z;
  const float* src = (z == 0) ? w0 : (z == 1) ? w1 : (z == 2) ? w2 : w3;
  u16* dst = (z == 0) ? o0 : (z == 1) ? o1 : (z == 2) ? o2 : o3;
  const size_t idx = ((size_t)blockIdx.x * 256 + threadIdx.x) * 4;
  fv4 f = *(const fv4*)(src + idx);
  unsigned long long pk = (unsigned long long)f2bf(f.x)
                        | ((unsigned long long)f2bf(f.y) << 16)
                        | ((unsigned long long)f2bf(f.z) << 32)
                        | ((unsigned long long)f2bf(f.w) << 48);
  *(unsigned long long*)(dst + idx) = pk;
}

// ---------------------------------------------------------------- RMSNorm->bf16
__global__ __launch_bounds__(256)
void rmsnorm_kernel(const float* __restrict__ x, const float* __restrict__ ln_w,
                    u16* __restrict__ xn) {
  const int row = blockIdx.x;
  const int tid = threadIdx.x;
  const float* xr = x + (size_t)row * D_DIM;
  float ss = 0.f;
  for (int c = tid; c < D_DIM; c += 256) { float v = xr[c]; ss += v * v; }
#pragma unroll
  for (int off = 16; off > 0; off >>= 1) ss += __shfl_xor(ss, off, 32);
  __shared__ float red[8];
  if ((tid & 31) == 0) red[tid >> 5] = ss;
  __syncthreads();
  float tot = 0.f;
#pragma unroll
  for (int i = 0; i < 8; ++i) tot += red[i];
  const float inv = rsqrtf(tot / (float)D_DIM + 1e-6f);
  u16* xo = xn + (size_t)row * D_DIM;
  for (int c = tid; c < D_DIM; c += 256) xo[c] = f2bf(ln_w[c] * xr[c] * inv);
}

// ------------------------------------------------- WMMA GEMM: C = A @ W^T (+R)
// A: [M,K] bf16 row-major, W: [N,K] bf16 row-major (both contiguous along K).
// TDM-fed double-buffered LDS tiles; wave 0 drives the DMA, TENSORcnt paces it.
// OUT_MODE 0: bf16 output. OUT_MODE 1: f32 output + residual add.
template<int OUT_MODE>
__global__ __launch_bounds__(256)
void gemm_wmma_kernel(const u16* __restrict__ A, const u16* __restrict__ W,
                      u16* __restrict__ outBf, float* __restrict__ outF,
                      const float* __restrict__ resid, int M, int N, int K) {
  __shared__ u16 As[2][128 * 32];
  __shared__ u16 Bs[2][128 * 32];
  const int tid   = threadIdx.x;
  const int lane  = tid & 31;
  const int wid   = tid >> 5;
  const int waveM = wid & 3;   // 4 waves along M (4*32 = 128)
  const int waveN = wid >> 2;  // 2 waves along N (2*64 = 128)
  const int m0 = blockIdx.y * 128;
  const int n0 = blockIdx.x * 128;
  (void)M;

  v8f acc[2][4];
#pragma unroll
  for (int mt = 0; mt < 2; ++mt)
#pragma unroll
    for (int nt = 0; nt < 4; ++nt)
      acc[mt][nt] = (v8f){0.f, 0.f, 0.f, 0.f, 0.f, 0.f, 0.f, 0.f};

  const int nIter = K / 32;
  if (wid == 0) {  // prologue: DMA tile 0 into buffer 0
    tdm_load_tile_2d(A + (size_t)m0 * K, (unsigned)(uintptr_t)&As[0][0], 32, 128, (unsigned)K);
    tdm_load_tile_2d(W + (size_t)n0 * K, (unsigned)(uintptr_t)&Bs[0][0], 32, 128, (unsigned)K);
  }

  for (int it = 0; it < nIter; ++it) {
    const int buf = it & 1;
    if (wid == 0) {
      if (it + 1 < nIter) {
        const int k1 = (it + 1) * 32;
        tdm_load_tile_2d(A + (size_t)m0 * K + k1, (unsigned)(uintptr_t)&As[buf ^ 1][0],
                         32, 128, (unsigned)K);
        tdm_load_tile_2d(W + (size_t)n0 * K + k1, (unsigned)(uintptr_t)&Bs[buf ^ 1][0],
                         32, 128, (unsigned)K);
        // 2 newer descriptors in flight; wait until only they remain ->
        // current tile's pair (in-order per wave) has landed in LDS.
        __builtin_amdgcn_s_wait_tensorcnt(2);
      } else {
        __builtin_amdgcn_s_wait_tensorcnt(0);
      }
    }
    __syncthreads();

    v16bf af[2], bfr[4];
#pragma unroll
    for (int mt = 0; mt < 2; ++mt)
      af[mt] = load_frag(&As[buf][0] + (waveM * 32 + mt * 16) * 32, 32, lane);
#pragma unroll
    for (int nt = 0; nt < 4; ++nt)
      bfr[nt] = load_frag(&Bs[buf][0] + (waveN * 64 + nt * 16) * 32, 32, lane);
#pragma unroll
    for (int mt = 0; mt < 2; ++mt)
#pragma unroll
      for (int nt = 0; nt < 4; ++nt)
        acc[mt][nt] = wmma_bf16(af[mt], bfr[nt], acc[mt][nt]);
    __syncthreads();
  }

  const int rowLo = (lane >> 4) * 8;  // C layout: lanes 16..31 hold rows M+8
  const int colL  = lane & 15;
#pragma unroll
  for (int mt = 0; mt < 2; ++mt)
#pragma unroll
    for (int nt = 0; nt < 4; ++nt)
#pragma unroll
      for (int r = 0; r < 8; ++r) {
        const int grow = m0 + waveM * 32 + mt * 16 + r + rowLo;
        const int gcol = n0 + waveN * 64 + nt * 16 + colL;
        const float v = acc[mt][nt][r];
        if (OUT_MODE == 0) {
          outBf[(size_t)grow * N + gcol] = f2bf(v);
        } else {
          outF[(size_t)grow * N + gcol] = v + resid[(size_t)grow * N + gcol];
        }
      }
}

// ---------------------------------------------------------------- T5 bias
__global__ __launch_bounds__(256)
void bias_kernel(const float* __restrict__ rel_emb, float* __restrict__ bias) {
  const int h = blockIdx.x >> 10;     // / S_LEN
  const int i = blockIdx.x & 1023;    // % S_LEN
  const int tid = threadIdx.x;
  float* brow = bias + ((size_t)h * S_LEN + i) * S_LEN;
  const float inv_log16_x8 = 8.0f / logf(16.0f);
#pragma unroll
  for (int e = 0; e < 4; ++e) {
    const int j = e * 256 + tid;
    const int rp = j - i;
    const int rb = (rp > 0) ? 16 : 0;
    const int rpa = (rp < 0) ? -rp : rp;
    int bkt;
    if (rpa < 8) {
      bkt = rpa;
    } else {
      int lg = 8 + (int)(logf((float)rpa * 0.125f) * inv_log16_x8);
      bkt = (lg < 15) ? lg : 15;
    }
    brow[j] = rel_emb[(rb + bkt) * H_NUM + h];
  }
}

// ------------------------------------------------- flash attention with WMMA
// Block = 8 waves, each wave owns 16 q rows; K/V tiles (32 kv rows) shared via
// LDS; online softmax in fp32 on WMMA C-layout; P round-trips through LDS.
__global__ __launch_bounds__(256)
void attention_kernel(const u16* __restrict__ Q, const u16* __restrict__ Kg,
                      const u16* __restrict__ Vg, const float* __restrict__ bias,
                      u16* __restrict__ AO) {
  __shared__ u16 Kt[32 * 64];       // [kv][hd]
  __shared__ u16 Vt[64 * 32];       // [hd][kv] (transposed on store)
  __shared__ u16 Ps[8 * 16 * 32];   // per-wave P scratch [16 q][32 kv]
  const int tid  = threadIdx.x;
  const int lane = tid & 31;
  const int wid  = tid >> 5;
  const int b = blockIdx.z;
  const int h = blockIdx.y;
  const int q0 = blockIdx.x * 128 + wid * 16;
  const size_t rowBase = (size_t)b * S_LEN;
  const int rowLo = (lane >> 4) * 8;
  const int colL  = lane & 15;

  const v16bf qa0 = load_frag(Q + (rowBase + q0) * D_DIM + h * HD,      D_DIM, lane);
  const v16bf qa1 = load_frag(Q + (rowBase + q0) * D_DIM + h * HD + 32, D_DIM, lane);

  v8f o[4];
#pragma unroll
  for (int nt = 0; nt < 4; ++nt) o[nt] = (v8f){0.f, 0.f, 0.f, 0.f, 0.f, 0.f, 0.f, 0.f};
  float m[8], l[8];
#pragma unroll
  for (int r = 0; r < 8; ++r) { m[r] = -__builtin_inff(); l[r] = 0.f; }

  const int ldr = tid >> 3;        // kv row 0..31 for cooperative load
  const int ldc = (tid & 7) * 8;   // hd col 0,8,..,56 (8 bf16 = 16 B)
  u16* pw = Ps + wid * (16 * 32);

  for (int kv0 = 0; kv0 < S_LEN; kv0 += 32) {
    uv4 kv4 = *(const uv4*)(Kg + (rowBase + kv0 + ldr) * D_DIM + h * HD + ldc);
    *(uv4*)(Kt + ldr * 64 + ldc) = kv4;
    uv4 vv4 = *(const uv4*)(Vg + (rowBase + kv0 + ldr) * D_DIM + h * HD + ldc);
    const u16* vs = (const u16*)&vv4;
#pragma unroll
    for (int e = 0; e < 8; ++e) Vt[(ldc + e) * 32 + ldr] = vs[e];
    __syncthreads();

    // scores: q-tile (16 x 64) x K^T -> two 16x16 fragments (kv halves)
    const v16bf kb00 = load_frag(Kt + 0  * 64 + 0,  64, lane);
    const v16bf kb01 = load_frag(Kt + 0  * 64 + 32, 64, lane);
    const v16bf kb10 = load_frag(Kt + 16 * 64 + 0,  64, lane);
    const v16bf kb11 = load_frag(Kt + 16 * 64 + 32, 64, lane);
    const v8f z = (v8f){0.f, 0.f, 0.f, 0.f, 0.f, 0.f, 0.f, 0.f};
    v8f s0 = wmma_bf16(qa1, kb01, wmma_bf16(qa0, kb00, z));
    v8f s1 = wmma_bf16(qa1, kb11, wmma_bf16(qa0, kb10, z));
    // NOTE: reference's k *= sqrt(D/H) cancels 1/sqrt(hd) -> no scaling here.

#pragma unroll
    for (int r = 0; r < 8; ++r) {
      const int qr = q0 + r + rowLo;
      const float* bp = bias + ((size_t)h * S_LEN + qr) * S_LEN + kv0 + colL;
      const float t0 = s0[r] + bp[0];
      const float t1 = s1[r] + bp[16];
      float mt = fmaxf(t0, t1);
#pragma unroll
      for (int off = 8; off > 0; off >>= 1) mt = fmaxf(mt, __shfl_xor(mt, off, 16));
      const float mn = fmaxf(m[r], mt);
      const float sc = __expf(m[r] - mn);
      const float p0 = __expf(t0 - mn);
      const float p1 = __expf(t1 - mn);
      float rs = p0 + p1;
#pragma unroll
      for (int off = 8; off > 0; off >>= 1) rs += __shfl_xor(rs, off, 16);
      l[r] = l[r] * sc + rs;
      m[r] = mn;
#pragma unroll
      for (int nt = 0; nt < 4; ++nt) o[nt][r] *= sc;
      pw[(r + rowLo) * 32 + colL]      = f2bf(p0);
      pw[(r + rowLo) * 32 + colL + 16] = f2bf(p1);
    }

    // O += P (16x32) @ V (32x64)
    const v16bf pa = load_frag(pw, 32, lane);
#pragma unroll
    for (int nt = 0; nt < 4; ++nt) {
      const v16bf vb = load_frag(Vt + nt * 16 * 32, 32, lane);
      o[nt] = wmma_bf16(pa, vb, o[nt]);
    }
    __syncthreads();
  }

#pragma unroll
  for (int nt = 0; nt < 4; ++nt)
#pragma unroll
    for (int r = 0; r < 8; ++r) {
      const int grow = q0 + r + rowLo;
      AO[(rowBase + grow) * D_DIM + h * HD + nt * 16 + colL] = f2bf(o[nt][r] / l[r]);
    }
}

// ---------------------------------------------------------------- launcher
extern "C" void kernel_launch(void* const* d_in, const int* in_sizes, int n_in,
                              void* d_out, int out_size, void* d_ws, size_t ws_size,
                              hipStream_t stream) {
  (void)in_sizes; (void)n_in; (void)out_size; (void)ws_size;
  const float* x   = (const float*)d_in[0];
  const float* wq  = (const float*)d_in[1];
  const float* wk  = (const float*)d_in[2];
  const float* wv  = (const float*)d_in[3];
  const float* wo  = (const float*)d_in[4];
  const float* lnw = (const float*)d_in[5];
  const float* rel = (const float*)d_in[6];

  float* out  = (float*)d_out;
  float* bias = out + (size_t)B_NUM * S_LEN * D_DIM;  // tuple elem 2: [1,H,S,S]

  u16* ws16 = (u16*)d_ws;
  const size_t XN = (size_t)B_NUM * S_LEN * D_DIM;  // 8 388 608
  const size_t WN = (size_t)D_DIM * D_DIM;          // 16 777 216
  u16* xnb = ws16;
  u16* wqb = xnb + XN;
  u16* wkb = wqb + WN;
  u16* wvb = wkb + WN;
  u16* wob = wvb + WN;
  u16* qb  = wob + WN;
  u16* kb  = qb + XN;
  u16* vb  = kb + XN;
  u16* aob = vb + XN;

  cast_weights_kernel<<<dim3((unsigned)(WN / 1024), 1, 4), 256, 0, stream>>>(
      wq, wk, wv, wo, wqb, wkb, wvb, wob);
  rmsnorm_kernel<<<dim3((unsigned)(XN / D_DIM)), 256, 0, stream>>>(x, lnw, xnb);

  dim3 gg(D_DIM / 128, (B_NUM * S_LEN) / 128);
  gemm_wmma_kernel<0><<<gg, 256, 0, stream>>>(xnb, wqb, qb, nullptr, nullptr, 2048, 4096, 4096);
  gemm_wmma_kernel<0><<<gg, 256, 0, stream>>>(xnb, wkb, kb, nullptr, nullptr, 2048, 4096, 4096);
  gemm_wmma_kernel<0><<<gg, 256, 0, stream>>>(xnb, wvb, vb, nullptr, nullptr, 2048, 4096, 4096);

  bias_kernel<<<dim3(H_NUM * S_LEN), 256, 0, stream>>>(rel, bias);

  attention_kernel<<<dim3(S_LEN / 128, H_NUM, B_NUM), 256, 0, stream>>>(qb, kb, vb, bias, aob);

  gemm_wmma_kernel<1><<<gg, 256, 0, stream>>>(aob, wob, nullptr, out, x, 2048, 4096, 4096);
}
